// InterventionModel_91044716740975
// MI455X (gfx1250) — matrix operand order repaired and checked by reading, not compiled, not run on hardware
//
#include <hip/hip_runtime.h>

// ---------------------------------------------------------------------------
// Fused per-(s,a)-expert 3-layer MLP with masked accumulation over action
// types, out = state_embed + sum_a mask[s,a]*relu(MLP_{s,a}(action[:,a])).
// f16 WMMA (v_wmma_f32_16x16x32_f16) with fp32 accumulate, weights staged
// K-major in LDS, all three layers fused (no [B,S,A,H] intermediates).
// ---------------------------------------------------------------------------

typedef __attribute__((ext_vector_type(16))) _Float16     v16h;
typedef __attribute__((ext_vector_type(8)))  float        v8f;
typedef __attribute__((ext_vector_type(4)))  unsigned int v4u;

#define B_  2048
#define S_  16
#define A_  16
#define DS_ 128
#define DA_ 128
#define H_  64

// LDS row strides (in halves), padded +8 to break 64-bank alignment.
#define LDW1 (DA_ + 8)   // sW1[h][d]   : 64 x 136
#define LDW2 (H_  + 8)   // sW2[n][k]   : 64 x 72
#define LDW3 (H_  + 8)   // sW3[e][k]   : 128 x 72
#define LDH  (H_  + 8)   // sH [m][k]   : 16 x 72 per wave

// A-matrix fragment (16x32 f16): per lane (r = lane&15, hi = lane>>4),
// elements 0..7  = K = kb + hi*8 + {0..7}
// elements 8..15 = K = kb + 16 + hi*8 + {0..7}
// -> two 16B loads with a 16-half gap.
static __device__ inline v16h frag_a_lds(const _Float16* p0) {
  union { v4u u[2]; v16h h; } u;
  u.u[0] = *(const v4u*)(p0);
  u.u[1] = *(const v4u*)(p0 + 16);
  return u.h;
}

// Same A-fragment pattern but sourced from fp32 global memory (cvt to f16).
static __device__ inline v16h frag_a_f32(const float* p0) {
  v16h r;
#pragma unroll
  for (int i = 0; i < 8; ++i) r[i]     = (_Float16)p0[i];
#pragma unroll
  for (int i = 0; i < 8; ++i) r[8 + i] = (_Float16)p0[16 + i];
  return r;
}

// B-matrix fragment (32x16 f16), K-major LDS: per lane (n = lane&15,
// hi = lane>>4), elements 0..7 = K = kb + hi*8 + {0..7}, elements 8..15 =
// K = kb + 16 + hi*8 + {0..7} -> two 16B loads with a 16-half gap.
static __device__ inline v16h frag_b_lds(const _Float16* p0) {
  union { v4u u[2]; v16h h; } u;
  u.u[0] = *(const v4u*)(p0);
  u.u[1] = *(const v4u*)(p0 + 16);
  return u.h;
}

__launch_bounds__(256, 1)
__global__ void intervention_kernel(const float* __restrict__ state,    // [B,S,DS]
                                    const float* __restrict__ action,   // [B,A,DA]
                                    const unsigned char* __restrict__ mask, // [S,A] bool
                                    const float* __restrict__ W1,       // [S,A,DA,H]
                                    const float* __restrict__ b1,       // [S,A,H]
                                    const float* __restrict__ W2,       // [S,A,H,H]
                                    const float* __restrict__ b2,       // [S,A,H]
                                    const float* __restrict__ W3,       // [S,A,H,DS]
                                    const float* __restrict__ b3,       // [S,A,DS]
                                    float* __restrict__ out)            // [B,S,DS]
{
  __shared__ __align__(16) _Float16 sW1[H_  * LDW1];   // 17408 B
  __shared__ __align__(16) _Float16 sW2[H_  * LDW2];   //  9216 B
  __shared__ __align__(16) _Float16 sW3[DS_ * LDW3];   // 18432 B
  __shared__ __align__(16) _Float16 sH[8][16 * LDH];   // 18432 B (per-wave h1/h2)

  const int tid  = threadIdx.x;
  const int lane = tid & 31;
  const int wave = tid >> 5;
  const int s    = blockIdx.y;
  const int b0   = blockIdx.x * 128 + wave * 16;   // this wave's 16 batch rows

  const int row = lane & 15;   // A-row / B,C,D column index
  const int hi  = lane >> 4;   // lane half select

  _Float16* myH = sH[wave];

  // diff accumulator: 16x128 f32 tile, C-layout (VGPR v <-> row v+8*hi).
  v8f acc[8];
#pragma unroll
  for (int nt = 0; nt < 8; ++nt)
#pragma unroll
    for (int v = 0; v < 8; ++v) acc[nt][v] = 0.0f;

  for (int a = 0; a < A_; ++a) {
    int mraw = mask[s * A_ + a];
    if (__builtin_amdgcn_readfirstlane(mraw) == 0) continue;  // uniform skip
    const int sa = s * A_ + a;

    // -------- stage weights for expert (s,a) into LDS, K-major, f32->f16 ---
    for (int idx = tid; idx < DA_ * H_; idx += 256) {          // W1[d][h] -> sW1[h][d]
      int d = idx >> 6, h = idx & 63;
      sW1[h * LDW1 + d] = (_Float16)W1[((size_t)sa * DA_ + d) * H_ + h];
    }
    for (int idx = tid; idx < H_ * H_; idx += 256) {           // W2[k][n] -> sW2[n][k]
      int k = idx >> 6, n = idx & 63;
      sW2[n * LDW2 + k] = (_Float16)W2[((size_t)sa * H_ + k) * H_ + n];
    }
    for (int idx = tid; idx < H_ * DS_; idx += 256) {          // W3[k][e] -> sW3[e][k]
      int k = idx >> 7, e = idx & 127;
      sW3[e * LDW3 + k] = (_Float16)W3[((size_t)sa * H_ + k) * DS_ + e];
    }
    __syncthreads();

    // -------- GEMM1: h1 = relu(act[16x128] @ W1[128x64] + b1) --------------
    const float* actp = action + ((size_t)(b0 + row) * A_ + a) * DA_;
    v16h a1[4];
#pragma unroll
    for (int ks = 0; ks < 4; ++ks)
      a1[ks] = frag_a_f32(actp + ks * 32 + hi * 8);

#pragma unroll
    for (int nt = 0; nt < 4; ++nt) {
      float bias = b1[(size_t)sa * H_ + nt * 16 + row];
      v8f c;
#pragma unroll
      for (int v = 0; v < 8; ++v) c[v] = bias;
#pragma unroll
      for (int ks = 0; ks < 4; ++ks) {
        v16h bf = frag_b_lds(sW1 + (nt * 16 + row) * LDW1 + ks * 32 + hi * 8);
        c = __builtin_amdgcn_wmma_f32_16x16x32_f16(false, a1[ks], false, bf,
                                                   (short)0, c, false, false);
      }
#pragma unroll
      for (int v = 0; v < 8; ++v) {
        float x = c[v]; x = x > 0.0f ? x : 0.0f;
        myH[(v + hi * 8) * LDH + nt * 16 + row] = (_Float16)x;  // h1 -> LDS (f16)
      }
    }

    // -------- GEMM2: h2 = relu(h1[16x64] @ W2[64x64] + b2) -----------------
    v16h a2[2];
#pragma unroll
    for (int ks = 0; ks < 2; ++ks)
      a2[ks] = frag_a_lds(myH + row * LDH + ks * 32 + hi * 8);

    v8f c2[4];
#pragma unroll
    for (int nt = 0; nt < 4; ++nt) {
      float bias = b2[(size_t)sa * H_ + nt * 16 + row];
#pragma unroll
      for (int v = 0; v < 8; ++v) c2[nt][v] = bias;
#pragma unroll
      for (int ks = 0; ks < 2; ++ks) {
        v16h bf = frag_b_lds(sW2 + (nt * 16 + row) * LDW2 + ks * 32 + hi * 8);
        c2[nt] = __builtin_amdgcn_wmma_f32_16x16x32_f16(false, a2[ks], false, bf,
                                                        (short)0, c2[nt], false, false);
      }
    }
    // all h1 reads done -> overwrite per-wave buffer with h2
#pragma unroll
    for (int nt = 0; nt < 4; ++nt)
#pragma unroll
      for (int v = 0; v < 8; ++v) {
        float x = c2[nt][v]; x = x > 0.0f ? x : 0.0f;
        myH[(v + hi * 8) * LDH + nt * 16 + row] = (_Float16)x;
      }

    // -------- GEMM3: acc += relu(h2[16x64] @ W3[64x128] + b3) --------------
    v16h a3[2];
#pragma unroll
    for (int ks = 0; ks < 2; ++ks)
      a3[ks] = frag_a_lds(myH + row * LDH + ks * 32 + hi * 8);

#pragma unroll
    for (int nt = 0; nt < 8; ++nt) {
      float bias = b3[(size_t)sa * DS_ + nt * 16 + row];
      v8f c;
#pragma unroll
      for (int v = 0; v < 8; ++v) c[v] = bias;
#pragma unroll
      for (int ks = 0; ks < 2; ++ks) {
        v16h bf = frag_b_lds(sW3 + (nt * 16 + row) * LDW3 + ks * 32 + hi * 8);
        c = __builtin_amdgcn_wmma_f32_16x16x32_f16(false, a3[ks], false, bf,
                                                   (short)0, c, false, false);
      }
#pragma unroll
      for (int v = 0; v < 8; ++v) {
        float x = c[v]; x = x > 0.0f ? x : 0.0f;
        acc[nt][v] += x;
      }
    }
    __syncthreads();  // weights consumed; safe to refill next iteration
  }

  // -------- out = state_embed + diff --------------------------------------
#pragma unroll
  for (int nt = 0; nt < 8; ++nt) {
    int col = nt * 16 + row;
#pragma unroll
    for (int v = 0; v < 8; ++v) {
      int b = b0 + v + hi * 8;
      size_t o = ((size_t)b * S_ + s) * DS_ + col;
      out[o] = state[o] + acc[nt][v];
    }
  }
}

extern "C" void kernel_launch(void* const* d_in, const int* in_sizes, int n_in,
                              void* d_out, int out_size, void* d_ws, size_t ws_size,
                              hipStream_t stream) {
  (void)in_sizes; (void)n_in; (void)out_size; (void)d_ws; (void)ws_size;
  const float*         state  = (const float*)d_in[0];
  const float*         action = (const float*)d_in[1];
  const unsigned char* mask   = (const unsigned char*)d_in[2];  // jax bool: 1 byte/elem
  const float*         W1     = (const float*)d_in[3];
  const float*         b1     = (const float*)d_in[4];
  const float*         W2     = (const float*)d_in[5];
  const float*         b2     = (const float*)d_in[6];
  const float*         W3     = (const float*)d_in[7];
  const float*         b3     = (const float*)d_in[8];
  float*               out    = (float*)d_out;

  dim3 grid(B_ / 128, S_, 1);   // 16 x 16 blocks, 8 waves each
  intervention_kernel<<<grid, 256, 0, stream>>>(state, action, mask,
                                                W1, b1, W2, b2, W3, b3, out);
}
